// ContextSelfAttn_12713103196256
// MI455X (gfx1250) — matrix operand order repaired and checked by reading, not compiled, not run on hardware
//
#include <hip/hip_runtime.h>
#include <hip/hip_bf16.h>
#include <stdint.h>

typedef __bf16 bf16_t;
typedef __attribute__((ext_vector_type(16))) __bf16 v16bf;
typedef __attribute__((ext_vector_type(8)))  __bf16 v8bf;
typedef __attribute__((ext_vector_type(8)))  float  v8f;
typedef __attribute__((ext_vector_type(4)))  unsigned int v4u;
typedef __attribute__((ext_vector_type(8)))  int v8i;
typedef __attribute__((ext_vector_type(4)))  int v4i;

#define BATCH 2
#define SEQ   2048
#define DMODEL 1024
#define NHEAD 16
#define HDIM  64
#define KROW  136   // 128 bf16 payload + 4-dword TDM pad (bank spread + 16B align)

__device__ inline v16bf pack16(const bf16_t* p0, const bf16_t* p1) {
    union { v16bf v; v8bf h[2]; } u;
    u.h[0] = *(const v8bf*)p0;
    u.h[1] = *(const v8bf*)p1;
    return u.v;
}

// ---------------------------------------------------------------------------
// Kernel 1: qk = enc @ W^T + bias   (M=4096, N=2048, K=1024), bf16 WMMA.
// Stores q  -> [bh][t][hd]      (bf16)
//        kT -> [bh][hd][s]*1/8  (bf16, transposed + prescaled)
// ---------------------------------------------------------------------------
__global__ __launch_bounds__(256) void proj_qk_kernel(
    const float* __restrict__ enc,   // [B*S, D]
    const float* __restrict__ W,     // [2D, D]
    const float* __restrict__ bias,  // [2D]
    bf16_t* __restrict__ qout,       // [B*H][S][HD]
    bf16_t* __restrict__ kTout)      // [B*H][HD][S]
{
    __shared__ __bf16 wtile[32][40];   // [e][d], padded for 16B-aligned v8bf reads

    const int tid  = threadIdx.x;
    const int lane = tid & 31;
    const int wid  = tid >> 5;
    const int mw   = wid & 3;
    const int nw   = wid >> 2;
    const int m0   = blockIdx.x * 64;  // row = b*S + s
    const int n0   = blockIdx.y * 32;  // col = e
    const int mrow = m0 + mw * 16 + (lane & 15);
    const int khalf = (lane >> 4) * 8; // ISA 16-bit A layout

    v8f acc = {};

    for (int k0 = 0; k0 < DMODEL; k0 += 32) {
        {   // cooperative W[n0..+32][k0..+32] f32 -> wtile[e][d] bf16 (same [n][k] layout as W)
            const int idx = tid * 4;
            const int e   = idx >> 5;
            const int d   = idx & 31;
            const float4 wv = *(const float4*)(W + (size_t)(n0 + e) * DMODEL + k0 + d);
            wtile[e][d + 0] = (__bf16)wv.x;
            wtile[e][d + 1] = (__bf16)wv.y;
            wtile[e][d + 2] = (__bf16)wv.z;
            wtile[e][d + 3] = (__bf16)wv.w;
        }
        __syncthreads();

        const float* ap = enc + (size_t)mrow * DMODEL + k0;
        __builtin_prefetch(ap + 32, 0, 1);
        const float4 a0 = *(const float4*)(ap + khalf);
        const float4 a1 = *(const float4*)(ap + khalf + 4);
        const float4 a2 = *(const float4*)(ap + 16 + khalf);
        const float4 a3 = *(const float4*)(ap + 16 + khalf + 4);
        v16bf av;
        av[0]=(__bf16)a0.x;  av[1]=(__bf16)a0.y;  av[2]=(__bf16)a0.z;  av[3]=(__bf16)a0.w;
        av[4]=(__bf16)a1.x;  av[5]=(__bf16)a1.y;  av[6]=(__bf16)a1.z;  av[7]=(__bf16)a1.w;
        av[8]=(__bf16)a2.x;  av[9]=(__bf16)a2.y;  av[10]=(__bf16)a2.z; av[11]=(__bf16)a2.w;
        av[12]=(__bf16)a3.x; av[13]=(__bf16)a3.y; av[14]=(__bf16)a3.z; av[15]=(__bf16)a3.w;

        union { v16bf v; v8bf h[2]; } ub;
        ub.h[0] = *(const v8bf*)&wtile[nw * 16 + (lane & 15)][khalf];
        ub.h[1] = *(const v8bf*)&wtile[nw * 16 + (lane & 15)][16 + khalf];

        acc = __builtin_amdgcn_wmma_f32_16x16x32_bf16(
                  false, av, false, ub.v, (short)0, acc, false, false);
        __syncthreads();
    }

    const int ncol  = n0 + nw * 16 + (lane & 15);
    const float bv  = bias[ncol];
    const int mbase = m0 + mw * 16 + 8 * (lane >> 4);
#pragma unroll
    for (int i = 0; i < 8; ++i) {
        const int m = mbase + i;
        const int b = m >> 11;
        const int s = m & (SEQ - 1);
        const float v = acc[i] + bv;
        if (ncol < DMODEL) {
            const int h = ncol >> 6, dd = ncol & 63;
            qout[((size_t)(b * NHEAD + h) * SEQ + s) * HDIM + dd] = (__bf16)v;
        } else {
            const int e2 = ncol - DMODEL;
            const int h = e2 >> 6, dd = e2 & 63;
            kTout[((size_t)(b * NHEAD + h) * HDIM + dd) * SEQ + s] = (__bf16)(v * 0.125f);
        }
    }
}

// ---------------------------------------------------------------------------
// Kernel 2: per (b,h, 16-row tile): scores = q . kT (causal) -> softmax -> out.
// K^T panels (64 x 128 bf16) are DMA'd HBM->LDS by the Tensor Data Mover,
// double-buffered, with TDM row padding (4 dw per 64 dw) giving KROW=136.
// ---------------------------------------------------------------------------
__global__ __launch_bounds__(256) void attn_softmax_kernel(
    const bf16_t* __restrict__ qmat,  // [BH][S][HD]
    const bf16_t* __restrict__ kT,    // [BH][HD][S]
    float* __restrict__ out)          // [BH][S][S]
{
    __shared__ float sc[16][SEQ + 4];
    __shared__ __align__(16) bf16_t ktile[2][HDIM * KROW];

    const int tid  = threadIdx.x;
    const int lane = tid & 31;
    const int wid  = tid >> 5;
    const int mt   = blockIdx.x & 127;
    const int bh   = blockIdx.x >> 7;
    const int t0   = mt * 16;
    const int npanels = (mt >> 3) + 1;   // panels of 8 column-tiles (128 cols)

    // Q fragments (A operand), hd split into two K=32 chunks
    const bf16_t* qp = qmat + ((size_t)bh * SEQ + t0 + (lane & 15)) * HDIM;
    const int khalf = (lane >> 4) * 8;
    const v16bf qa0 = pack16(qp + khalf,      qp + 16 + khalf);
    const v16bf qa1 = pack16(qp + 32 + khalf, qp + 48 + khalf);

    const uint64_t kgbase = (uint64_t)(uintptr_t)(kT + (size_t)bh * HDIM * SEQ);
    const int nloc = lane & 15;
    const int mb   = 8 * (lane >> 4);

    // --- TDM descriptor issue: 64x128 bf16 tile of the [64][2048] kT tensor ---
    auto issue_tdm = [&](int p, int buf) {
        const uint64_t ga  = kgbase + (uint64_t)(p * 128) * sizeof(bf16_t);
        const uint32_t lds = (uint32_t)(uintptr_t)&ktile[buf][0]; // flat low 32 = LDS offset
        v4u g0;
        g0[0] = 1u;                                   // count=1, no gather
        g0[1] = lds;                                  // lds_addr
        g0[2] = (unsigned int)ga;                     // global_addr[31:0]
        g0[3] = (unsigned int)((ga >> 32) & 0x1FFFFFFu) | (2u << 30); // addr[56:32] | type=2
        v8i g1;
        g1[0] = (1 << 16)            // data_size = 2 bytes
              | (1 << 20)            // pad_enable
              | (5 << 22)            // pad_interval: every 64 dwords
              | (3 << 25);           // pad_amount: 4 dwords -> row stride 136 bf16
        g1[1] = (int)((SEQ & 0xFFFF) << 16);          // tensor_dim0 lo16 (hi bits: atomic addr=0)
        g1[2] = (int)(((SEQ >> 16) & 0xFFFF) | (HDIM << 16)); // tensor_dim0 hi | tensor_dim1 lo
        g1[3] = (int)(128 << 16);                     // tensor_dim1 hi=0 | tile_dim0=128
        g1[4] = HDIM;                                 // tile_dim1=64 | tile_dim2=0
        g1[5] = SEQ;                                  // tensor_dim0_stride lo32
        g1[6] = 0;                                    // stride hi | dim1_stride lo (2D: unused)
        g1[7] = 0;
        const v4i z4 = {0, 0, 0, 0};                  // groups 2/3: 2D tensor
        const v8i z8 = {0, 0, 0, 0, 0, 0, 0, 0};
        __builtin_amdgcn_tensor_load_to_lds(g0, g1, z4, z4, z8, 0);
    };

    if (wid == 0) issue_tdm(0, 0);

    for (int p = 0; p < npanels; ++p) {
        if (wid == 0) {
            if (p + 1 < npanels) {   // overlap: start next panel, then wait for current
                issue_tdm(p + 1, (p + 1) & 1);
                __builtin_amdgcn_s_wait_tensorcnt(1);
            } else {
                __builtin_amdgcn_s_wait_tensorcnt(0);
            }
        }
        __syncthreads();             // publish buffer p&1 to all waves

        const int nt = p * 8 + wid;  // wave-uniform guard: EXEC stays all-ones
        if (nt <= mt) {
            const bf16_t* kp = &ktile[p & 1][wid * 16 + nloc];
            v16bf kb0, kb1;
#pragma unroll
            for (int s = 0; s < 8; ++s) {
                kb0[s]     = kp[(size_t)(khalf + s) * KROW];
                kb0[8 + s] = kp[(size_t)(16 + khalf + s) * KROW];
                kb1[s]     = kp[(size_t)(32 + khalf + s) * KROW];
                kb1[8 + s] = kp[(size_t)(48 + khalf + s) * KROW];
            }

            v8f c = {};
            c = __builtin_amdgcn_wmma_f32_16x16x32_bf16(false, qa0, false, kb0, (short)0, c, false, false);
            c = __builtin_amdgcn_wmma_f32_16x16x32_bf16(false, qa1, false, kb1, (short)0, c, false, false);

            const int s0 = nt * 16;
            if (nt == mt) {          // causal mask on diagonal tile
#pragma unroll
                for (int i = 0; i < 8; ++i)
                    if (nloc > mb + i) c[i] += -10000.0f;
            }
#pragma unroll
            for (int i = 0; i < 8; ++i)
                sc[mb + i][s0 + nloc] = c[i];
        }
        __syncthreads();             // all reads of buffer done before TDM reuses it
    }

    // softmax: 2 rows per wave
    for (int r = wid; r < 16; r += 8) {
        const int t = t0 + r;
        float mx = -3.0e38f;
        for (int j = lane; j <= t; j += 32) mx = fmaxf(mx, sc[r][j]);
#pragma unroll
        for (int off = 16; off; off >>= 1) mx = fmaxf(mx, __shfl_xor(mx, off, 32));

        float sum = 0.0f;
        for (int j = lane; j <= t; j += 32) sum += __expf(sc[r][j] - mx);
#pragma unroll
        for (int off = 16; off; off >>= 1) sum += __shfl_xor(sum, off, 32);

        const float inv = 1.0f / sum;
        float* orow = out + ((size_t)bh * SEQ + t) * SEQ;
        for (int j = lane; j < SEQ; j += 32)
            orow[j] = (j <= t) ? __expf(sc[r][j] - mx) * inv : 0.0f;
    }
}

// ---------------------------------------------------------------------------
extern "C" void kernel_launch(void* const* d_in, const int* in_sizes, int n_in,
                              void* d_out, int out_size, void* d_ws, size_t ws_size,
                              hipStream_t stream) {
    const float* enc  = (const float*)d_in[0];  // [2,2048,1024]
    const float* W    = (const float*)d_in[1];  // [2048,1024]
    const float* bias = (const float*)d_in[2];  // [2048]
    float* out = (float*)d_out;                 // [2,16,2048,2048]

    bf16_t* q  = (bf16_t*)d_ws;                              // 8 MB
    bf16_t* kT = q + (size_t)BATCH * NHEAD * SEQ * HDIM;     // 8 MB

    proj_qk_kernel<<<dim3(64, 64), 256, 0, stream>>>(enc, W, bias, q, kT);
    attn_softmax_kernel<<<dim3(BATCH * NHEAD * (SEQ / 16)), 256, 0, stream>>>(q, kT, out);
}
